// QTALSTMCell_47442208751723
// MI455X (gfx1250) — matrix operand order repaired and checked by reading, not compiled, not run on hardware
//
#include <hip/hip_runtime.h>
#include <hip/hip_bf16.h>
#include <math.h>

// ---------------------------------------------------------------------------
// QT-LSTM cell, MI455X (gfx1250, wave32, WMMA + async LDS loads)
// B=16384, D=512, H=512, K=D+H=1024, 4H=2048
//   gates = [x|h] @ W^T + b   -> fused sigmoid/tanh/LSTM update -> (h_t, c_t)
// GEMM in f16 with fp32 accumulation via v_wmma_f32_16x16x32_f16.
// W tiles staged into LDS with global_load_async_to_lds_b128 (ASYNCcnt).
// ---------------------------------------------------------------------------

typedef __attribute__((ext_vector_type(16))) _Float16 v16h;
typedef __attribute__((ext_vector_type(8)))  float    v8f;

#define B_ROWS 16384
#define DIN    512
#define HID    512
#define KDIM   1024   // D + H
#define GATES4 2048   // 4*H
#define KCHUNK 256    // K per LDS-staged chunk
#define NKCH   (KDIM / KCHUNK)   // 4 chunks
#define NKS    (KCHUNK / 32)     // 8 WMMA K-steps per chunk

// Per-lane async copy: 16 bytes global -> LDS (tracked with ASYNCcnt).
__device__ __forceinline__ void async_ld_b128(unsigned lds_off, const void* gptr) {
    asm volatile("global_load_async_to_lds_b128 %0, %1, off"
                 :: "v"(lds_off), "v"((unsigned long long)(uintptr_t)gptr)
                 : "memory");
}
__device__ __forceinline__ void wait_async0() {
    asm volatile("s_wait_asynccnt 0x0" ::: "memory");
}

// -------------------- phase 0: per-row confidence scalars -------------------
__global__ void qtlstm_scalars_k(const float* __restrict__ conf_t,
                                 const float* __restrict__ conf_prev,
                                 const float* __restrict__ alpha,
                                 const float* __restrict__ lambda_p,
                                 float* __restrict__ sArr,
                                 float* __restrict__ dArr) {
    int i = blockIdx.x * blockDim.x + threadIdx.x;
    if (i >= B_ROWS) return;
    float conf_s = 0.8f * conf_prev[i] + 0.2f * conf_t[i];
    float ar     = fmaxf(alpha[0], 0.0f);
    float s      = expf(-ar * (1.0f - conf_s));
    float lam    = 1.0f / (1.0f + expf(-lambda_p[0]));
    sArr[i] = s;
    dArr[i] = 1.0f - lam * (1.0f - s);
}

// -------------------- phase 1: fp32 -> f16 conversions ----------------------
__global__ void qtlstm_cvtA_k(const float* __restrict__ x,
                              const float* __restrict__ h,
                              _Float16* __restrict__ A) {
    int i = blockIdx.x * blockDim.x + threadIdx.x;
    if (i >= B_ROWS * KDIM) return;
    int b = i >> 10;          // /1024
    int k = i & (KDIM - 1);
    float v = (k < DIN) ? x[(size_t)b * DIN + k]
                        : h[(size_t)b * HID + (k - DIN)];
    A[i] = (_Float16)v;
}

__global__ void qtlstm_cvtW_k(const float* __restrict__ W,
                              _Float16* __restrict__ Wh) {
    int i = blockIdx.x * blockDim.x + threadIdx.x;
    if (i >= GATES4 * KDIM) return;
    Wh[i] = (_Float16)W[i];
}

// -------------------- phase 2: fused WMMA GEMM + LSTM -----------------------
// Block = 256 threads = 8 waves. Block (nx, my):
//   rows  my*128 .. my*128+127  (wave w owns 16 rows)
//   cols  nx*16  .. nx*16+15    in EACH of the 4 gate slices (f,i,o,g)
// K in 4 chunks of 256; per chunk: async-stage 32KB W tile -> LDS (one wait +
// one barrier), then 8 barrier-free inner K-steps of ds_load + 4x WMMA.
__global__ __launch_bounds__(256)
void qtlstm_gemm_lstm_k(const _Float16* __restrict__ A,    // [B][1024] f16
                        const _Float16* __restrict__ Wh,   // [2048][1024] f16
                        const float*    __restrict__ bias, // [2048]
                        const float*    __restrict__ c_prev,
                        const float*    __restrict__ sArr,
                        const float*    __restrict__ dArr,
                        float*          __restrict__ out)  // h_t then c_t
{
    __shared__ __attribute__((aligned(64))) _Float16 Wt[4][16][KCHUNK]; // 32 KB

    const int t    = threadIdx.x;
    const int wave = t >> 5;
    const int lane = t & 31;
    const int col  = lane & 15;   // output column within tile (N)
    const int hi   = lane >> 4;   // lane-group select

    const int rowBase = blockIdx.y * 128 + wave * 16;
    const int n0      = blockIdx.x * 16;

    v8f accF = {}, accI = {}, accO = {}, accG = {};

    // A fragment source: row = rowBase + col, two 16B chunks per K-step
    // (contiguous K order, same as B: lanes 0-15 hold K {0..15}, lanes 16-31 {16..31})
    const uint4* arow = (const uint4*)(A + (size_t)(rowBase + col) * KDIM);

    // LDS base byte offset (hardware adds LDS_BASE; low 32 bits of generic
    // shared pointer are the workgroup-relative LDS address).
    const unsigned ldsWt = (unsigned)(uintptr_t)(void*)&Wt[0][0][0];

    for (int kc = 0; kc < NKCH; ++kc) {
        // -------- async-stage 32KB W tile: 2048 x 16B chunks, 8 per thread --
        // chunk c = q*256 + t : one wave covers one 512B row (32 lanes x 16B)
        #pragma unroll
        for (int q = 0; q < 8; ++q) {
            const int c  = q * 256 + t;
            const int r  = c >> 5;        // LDS row 0..63  (g*16 + n)
            const int cc = c & 31;        // 16B chunk within row
            const int g  = r >> 4;
            const int n  = r & 15;
            const _Float16* src = Wh + (size_t)(g * HID + n0 + n) * KDIM
                                     + kc * KCHUNK + cc * 8;
            async_ld_b128(ldsWt + (unsigned)(r * (KCHUNK * 2) + cc * 16), src);
        }
        wait_async0();       // this thread's async copies complete
        __syncthreads();     // all threads' copies complete -> tile ready

        // -------- 8 barrier-free WMMA K-steps over this chunk --------------
        #pragma unroll
        for (int ks = 0; ks < NKS; ++ks) {
            const int kstep = kc * NKS + ks;

            // prefetch next A chunk into cache
            __builtin_prefetch((const void*)(arow + kstep * 4 + 4), 0, 1);

            union { v16h v; uint4 q[2]; } a;
            a.q[0] = arow[kstep * 4 + hi * 2];       // K {0..7} or {16..23}
            a.q[1] = arow[kstep * 4 + hi * 2 + 1];   // K {8..15} or {24..31}

            v16h bF = *(const v16h*)&Wt[0][col][ks * 32 + hi * 16];
            v16h bI = *(const v16h*)&Wt[1][col][ks * 32 + hi * 16];
            v16h bO = *(const v16h*)&Wt[2][col][ks * 32 + hi * 16];
            v16h bG = *(const v16h*)&Wt[3][col][ks * 32 + hi * 16];

            accF = __builtin_amdgcn_wmma_f32_16x16x32_f16(false, a.v, false, bF,
                                                          (short)0, accF, false, false);
            accI = __builtin_amdgcn_wmma_f32_16x16x32_f16(false, a.v, false, bI,
                                                          (short)0, accI, false, false);
            accO = __builtin_amdgcn_wmma_f32_16x16x32_f16(false, a.v, false, bO,
                                                          (short)0, accO, false, false);
            accG = __builtin_amdgcn_wmma_f32_16x16x32_f16(false, a.v, false, bG,
                                                          (short)0, accG, false, false);
        }
        __syncthreads();     // tile fully consumed before restaging
    }

    // Bias: one scalar per lane per gate (lane column = n0 + col)
    const float bf = bias[0 * HID + n0 + col];
    const float bi = bias[1 * HID + n0 + col];
    const float bo = bias[2 * HID + n0 + col];
    const float bg = bias[3 * HID + n0 + col];
    const int   hc = n0 + col;

    // C/D layout: VGPR j -> row j (lanes 0-15) or j+8 (lanes 16-31)
    #pragma unroll
    for (int j = 0; j < 8; ++j) {
        const int m = rowBase + j + hi * 8;
        const float s   = sArr[m];
        const float dec = dArr[m];

        const float f  = 1.0f / (1.0f + expf(-(accF[j] + bf)));
        const float ii = (1.0f / (1.0f + expf(-(accI[j] + bi)))) * s;
        const float o  = 1.0f / (1.0f + expf(-(accO[j] + bo)));
        const float g  = tanhf(accG[j] + bg);

        const float cdec = c_prev[(size_t)m * HID + hc] * dec;
        const float c    = f * cdec + ii * g;
        const float ht   = o * tanhf(c);

        out[(size_t)m * HID + hc]                        = ht;  // h_t
        out[(size_t)B_ROWS * HID + (size_t)m * HID + hc] = c;   // c_t
    }
}

// ---------------------------------------------------------------------------
extern "C" void kernel_launch(void* const* d_in, const int* in_sizes, int n_in,
                              void* d_out, int out_size, void* d_ws, size_t ws_size,
                              hipStream_t stream) {
    const float* x        = (const float*)d_in[0];
    const float* h_prev   = (const float*)d_in[1];
    const float* c_prev   = (const float*)d_in[2];
    const float* conf_t   = (const float*)d_in[3];
    const float* conf_p   = (const float*)d_in[4];
    const float* W        = (const float*)d_in[5];
    const float* bias     = (const float*)d_in[6];
    const float* alpha    = (const float*)d_in[7];
    const float* lambda_p = (const float*)d_in[8];
    float*       out      = (float*)d_out;

    // workspace layout
    char* ws = (char*)d_ws;
    _Float16* Af16 = (_Float16*)ws;                                   // 32 MB
    _Float16* Wf16 = (_Float16*)(ws + (size_t)B_ROWS * KDIM * 2);     //  4 MB
    float*    sArr = (float*)(ws + (size_t)B_ROWS * KDIM * 2
                                 + (size_t)GATES4 * KDIM * 2);        // 64 KB
    float*    dArr = sArr + B_ROWS;                                   // 64 KB

    qtlstm_scalars_k<<<(B_ROWS + 255) / 256, 256, 0, stream>>>(
        conf_t, conf_p, alpha, lambda_p, sArr, dArr);

    qtlstm_cvtA_k<<<(B_ROWS * KDIM + 255) / 256, 256, 0, stream>>>(
        x, h_prev, Af16);

    qtlstm_cvtW_k<<<(GATES4 * KDIM + 255) / 256, 256, 0, stream>>>(
        W, Wf16);

    dim3 grid(HID / 16, B_ROWS / 128);   // (32, 128)
    qtlstm_gemm_lstm_k<<<grid, 256, 0, stream>>>(
        Af16, Wf16, bias, c_prev, sArr, dArr, out);
}
